// DeepPureSpectral_48808008351960
// MI455X (gfx1250) — matrix operand order repaired and checked by reading, not compiled, not run on hardware
//
#include <hip/hip_runtime.h>
#include <math.h>

// ---------------- types ----------------
typedef __attribute__((ext_vector_type(16))) _Float16 v16h;
typedef __attribute__((ext_vector_type(8)))  _Float16 v8h;
typedef __attribute__((ext_vector_type(4)))  _Float16 v4h;
typedef __attribute__((ext_vector_type(8)))  float    v8f;

#define BATCH 32768
#define INF   784
#define HF    256
#define OUTF  10
#define LPOW  100

// ---------------- fp32 -> f16 convert ----------------
__global__ void f32_to_f16_kernel(const float* __restrict__ in,
                                  _Float16* __restrict__ out, int n) {
    int i = blockIdx.x * blockDim.x + threadIdx.x;
    if (i < n) out[i] = (_Float16)in[i];
}

// ---------------- spectral collapse: build circulant (M^100) as f16 256x256 ----------------
// One step of the scan is h <- Re(IFFT(FFT(h)*w)) == circular conv with Re(IFFT(w)),
// a circulant with spectrum s[k] = (w[k] + conj(w[(N-k)%N]))/2. L steps => spectrum s^L.
// |s| ~ 0.01 so s^100 ~ 1e-200: must use fp64 for the power & inverse DFT.
__global__ __launch_bounds__(256)
void spectral_collapse_kernel(const float* __restrict__ wr,
                              const float* __restrict__ wi,
                              _Float16* __restrict__ Cmat) {
    __shared__ double pre[256], pim[256];
    __shared__ float  cbuf[256];
    const int k  = threadIdx.x;
    const int kr = (256 - k) & 255;
    const double sr = 0.5 * ((double)wr[k] + (double)wr[kr]);
    const double si = 0.5 * ((double)wi[k] - (double)wi[kr]);
    double pr = 1.0, pi = 0.0;
    for (int i = 0; i < LPOW; ++i) {
        double nr = pr * sr - pi * si;
        double ni = pr * si + pi * sr;
        pr = nr; pi = ni;
    }
    pre[k] = pr; pim[k] = pi;
    __syncthreads();
    // c[n] = (1/256) * sum_k Re(p[k] * exp(+i*2*pi*k*n/256))
    const int n = threadIdx.x;
    double accv = 0.0;
    for (int kk = 0; kk < 256; ++kk) {
        int t = (kk * n) & 255;
        double th = (double)t * (6.283185307179586476925286766559 / 256.0);
        accv += pre[kk] * cos(th) - pim[kk] * sin(th);
    }
    cbuf[n] = (float)(accv * (1.0 / 256.0));
    __syncthreads();
    // (h @ C)[b,n] = sum_j h[b,j]*c[(n-j) mod 256]  =>  C[j][n] = c[(n-j)&255]
    const int j = threadIdx.x;
    for (int nn = 0; nn < 256; ++nn)
        Cmat[j * 256 + nn] = (_Float16)cbuf[(nn - j) & 255];
}

// ---------------- WMMA GEMM: out[M x 256] = A[M x K] * Bw[K x 256] (+bias) ----------------
// BM=128, BN=256 (full width, A streamed once), BK=32. 8 waves/block, wave w owns
// rows [16w,16w+16), 16 n-tiles of 16x16 f32 accum (v_wmma_f32_16x16x32_f16).
#define BKK 32
#define LPAD 40  // pad LDS rows to 40 halfs (80B) to spread banks; keeps 16B alignment

template <typename TA>
__global__ __launch_bounds__(256)
void gemm_wmma_kernel(const TA* __restrict__ A, const _Float16* __restrict__ Bw,
                      const float* __restrict__ bias, float* __restrict__ Co,
                      int M, int K) {
    __shared__ alignas(16) _Float16 As[128][LPAD];
    __shared__ alignas(16) _Float16 Bs[256][LPAD];   // transposed: Bs[n][k]

    const int tid  = threadIdx.x;
    const int wave = tid >> 5;
    const int lane = tid & 31;
    const int lmod = lane & 15;
    const int half = lane >> 4;
    const int rowBase = blockIdx.x * 128;

    v8f acc[16] = {};

    const int KT = (K + BKK - 1) / BKK;
    for (int kt = 0; kt < KT; ++kt) {
        const int k0 = kt * BKK;
        // stage A tile 128x32 (convert to f16; zero-pad K remainder)
        #pragma unroll
        for (int i = 0; i < 16; ++i) {
            int idx = tid + i * 256;
            int r = idx >> 5;
            int c = idx & 31;
            float v = 0.f;
            if (k0 + c < K) v = (float)A[(size_t)(rowBase + r) * K + (k0 + c)];
            As[r][c] = (_Float16)v;
        }
        // stage B tile transposed: Bs[n][k], coalesced global reads of row-major B
        #pragma unroll
        for (int kk = 0; kk < BKK; ++kk) {
            _Float16 v = (_Float16)0.f;
            if (k0 + kk < K) v = Bw[(size_t)(k0 + kk) * 256 + tid];
            Bs[tid][kk] = v;
        }
        if (kt + 1 < KT)  // gfx1250 global_prefetch of next A tile
            __builtin_prefetch(&A[(size_t)(rowBase + (tid >> 1)) * K + (k0 + BKK)], 0, 1);
        __syncthreads();

        // A fragment (16-bit A 16x32 layout): lane m=L%16; k runs [8*half..) and [16+8*half..)
        const int am = wave * 16 + lmod;
        v8h alo = *(const v8h*)&As[am][half * 8];
        v8h ahi = *(const v8h*)&As[am][16 + half * 8];
        v16h a;
        #pragma unroll
        for (int i = 0; i < 8; ++i) { a[i] = alo[i]; a[i + 8] = ahi[i]; }

        #pragma unroll
        for (int nt = 0; nt < 16; ++nt) {
            // B fragment (16-bit B 32x16): lane n=L%16; k runs [8*half..) and [16+8*half..)
            v8h blo = *(const v8h*)&Bs[nt * 16 + lmod][half * 8];
            v8h bhi = *(const v8h*)&Bs[nt * 16 + lmod][16 + half * 8];
            v16h b;
            #pragma unroll
            for (int i = 0; i < 8; ++i) { b[i] = blo[i]; b[i + 8] = bhi[i]; }
            acc[nt] = __builtin_amdgcn_wmma_f32_16x16x32_f16(
                false, a, false, b, (short)0, acc[nt], false, false);
        }
        __syncthreads();
    }

    // epilogue: C/D layout — VGPR r: row = 16*wave + 8*half + r, col = nt*16 + L%16
    const int outRow0 = rowBase + wave * 16 + half * 8;
    #pragma unroll
    for (int nt = 0; nt < 16; ++nt) {
        const int col = nt * 16 + lmod;
        const float bv = bias ? bias[col] : 0.f;
        #pragma unroll
        for (int r = 0; r < 8; ++r)
            Co[(size_t)(outRow0 + r) * 256 + col] = acc[nt][r] + bv;
    }
}

// ---------------- batchnorm column stats (sum, sumsq) ----------------
__global__ __launch_bounds__(256)
void colstats_kernel(const float* __restrict__ T, float* __restrict__ sum,
                     float* __restrict__ sumsq) {
    const int j = threadIdx.x;
    const size_t r0 = (size_t)blockIdx.x * 128;
    float s = 0.f, q = 0.f;
    for (int r = 0; r < 128; ++r) {
        float v = T[(r0 + r) * 256 + j];
        s += v; q += v * v;
    }
    atomicAdd(&sum[j], s);
    atomicAdd(&sumsq[j], q);
}

__global__ void bn_finalize_kernel(const float* __restrict__ sum,
                                   const float* __restrict__ sumsq,
                                   const float* __restrict__ gamma,
                                   const float* __restrict__ beta,
                                   float* __restrict__ scale,
                                   float* __restrict__ shift) {
    const int j = threadIdx.x;
    const float invB = 1.0f / (float)BATCH;
    float mu  = sum[j] * invB;
    float var = fmaxf(sumsq[j] * invB - mu * mu, 0.f);
    float sc  = gamma[j] * rsqrtf(var + 1e-5f);
    scale[j] = sc;
    shift[j] = beta[j] - mu * sc;
}

__global__ void zero_stats_kernel(float* __restrict__ st) {
    for (int i = threadIdx.x; i < 2048; i += 256) st[i] = 0.f;
}

// ---------------- BN + ReLU + convert to f16 (fused) ----------------
__global__ __launch_bounds__(256)
void bn_relu_f16_kernel(const float* __restrict__ T, const float* __restrict__ scale,
                        const float* __restrict__ shift, _Float16* __restrict__ Hb) {
    __shared__ float sc[256], sh[256];
    const int tid = threadIdx.x;
    sc[tid] = scale[tid]; sh[tid] = shift[tid];
    __syncthreads();
    const size_t idx4 = (size_t)blockIdx.x * 256 + tid;
    const float4 v = ((const float4*)T)[idx4];
    const int j0 = (int)((idx4 * 4) & 255);
    v4h o;
    o[0] = (_Float16)fmaxf(v.x * sc[j0]     + sh[j0],     0.f);
    o[1] = (_Float16)fmaxf(v.y * sc[j0 + 1] + sh[j0 + 1], 0.f);
    o[2] = (_Float16)fmaxf(v.z * sc[j0 + 2] + sh[j0 + 2], 0.f);
    o[3] = (_Float16)fmaxf(v.w * sc[j0 + 3] + sh[j0 + 3], 0.f);
    *(v4h*)&Hb[idx4 * 4] = o;
}

// ---------------- BN2 + ReLU fused with tiny 256->10 projection ----------------
__global__ __launch_bounds__(256)
void bn_relu_gemm_out_kernel(const float* __restrict__ T2,
                             const float* __restrict__ scale,
                             const float* __restrict__ shift,
                             const float* __restrict__ W2,
                             const float* __restrict__ b2,
                             float* __restrict__ out) {
    __shared__ float w2s[256 * OUTF];
    __shared__ float sc[256], sh[256];
    const int tid = threadIdx.x;
    for (int i = tid; i < 256 * OUTF; i += 256) w2s[i] = W2[i];
    sc[tid] = scale[tid]; sh[tid] = shift[tid];
    __syncthreads();
    const size_t row = (size_t)blockIdx.x * 256 + tid;
    float acc[OUTF];
    #pragma unroll
    for (int o = 0; o < OUTF; ++o) acc[o] = b2[o];
    const float4* rp = (const float4*)&T2[row * 256];
    for (int k4 = 0; k4 < 64; ++k4) {
        float4 v = rp[k4];
        int k = k4 * 4;
        float h0 = fmaxf(v.x * sc[k]     + sh[k],     0.f);
        float h1 = fmaxf(v.y * sc[k + 1] + sh[k + 1], 0.f);
        float h2 = fmaxf(v.z * sc[k + 2] + sh[k + 2], 0.f);
        float h3 = fmaxf(v.w * sc[k + 3] + sh[k + 3], 0.f);
        #pragma unroll
        for (int o = 0; o < OUTF; ++o)
            acc[o] += h0 * w2s[k * OUTF + o] + h1 * w2s[(k + 1) * OUTF + o]
                    + h2 * w2s[(k + 2) * OUTF + o] + h3 * w2s[(k + 3) * OUTF + o];
    }
    #pragma unroll
    for (int o = 0; o < OUTF; ++o) out[row * OUTF + o] = acc[o];
}

// ---------------- launcher ----------------
extern "C" void kernel_launch(void* const* d_in, const int* in_sizes, int n_in,
                              void* d_out, int out_size, void* d_ws, size_t ws_size,
                              hipStream_t stream) {
    const float* x   = (const float*)d_in[0];
    const float* W1  = (const float*)d_in[1];
    const float* b1  = (const float*)d_in[2];
    const float* g1  = (const float*)d_in[3];
    const float* be1 = (const float*)d_in[4];
    const float* wr  = (const float*)d_in[5];
    const float* wi  = (const float*)d_in[6];
    const float* g2  = (const float*)d_in[7];
    const float* be2 = (const float*)d_in[8];
    const float* W2  = (const float*)d_in[9];
    const float* b2  = (const float*)d_in[10];
    float* out = (float*)d_out;

    char* ws = (char*)d_ws;
    float*    T   = (float*)(ws);                 // 32768*256 f32 (T1, reused as T2)
    _Float16* Hb  = (_Float16*)(ws + 33554432);   // 32768*256 f16
    _Float16* W1h = (_Float16*)(ws + 50331648);   // 784*256 f16
    _Float16* Cm  = (_Float16*)(ws + 50733056);   // 256*256 f16 (collapsed spectral op)
    float*    st  = (float*)(ws + 50864128);      // 2048 f32 stats/scale/shift

    f32_to_f16_kernel<<<784, 256, 0, stream>>>(W1, W1h, INF * HF);
    spectral_collapse_kernel<<<1, 256, 0, stream>>>(wr, wi, Cm);
    zero_stats_kernel<<<1, 256, 0, stream>>>(st);

    // h1 = x @ W1 + b1
    gemm_wmma_kernel<float><<<BATCH / 128, 256, 0, stream>>>(x, W1h, b1, T, BATCH, INF);
    // BN1 stats -> scale/shift, then apply BN+ReLU -> f16
    colstats_kernel<<<BATCH / 128, 256, 0, stream>>>(T, st, st + 256);
    bn_finalize_kernel<<<1, 256, 0, stream>>>(st, st + 256, g1, be1, st + 512, st + 768);
    bn_relu_f16_kernel<<<(BATCH * HF) / 1024, 256, 0, stream>>>(T, st + 512, st + 768, Hb);

    // 100 spectral steps collapsed: h2 = Hb @ C
    gemm_wmma_kernel<_Float16><<<BATCH / 128, 256, 0, stream>>>(Hb, Cm, nullptr, T, BATCH, HF);

    // BN2 stats, then fused BN+ReLU+(256->10 proj)+bias
    colstats_kernel<<<BATCH / 128, 256, 0, stream>>>(T, st + 1024, st + 1280);
    bn_finalize_kernel<<<1, 256, 0, stream>>>(st + 1024, st + 1280, g2, be2, st + 1536, st + 1792);
    bn_relu_gemm_out_kernel<<<BATCH / 256, 256, 0, stream>>>(T, st + 1536, st + 1792, W2, b2, out);
}
